// SonModel_60739427500666
// MI455X (gfx1250) — compile-verified, matrix operand order checked by reading, not run on hardware
//
#include <hip/hip_runtime.h>
#include <hip/hip_bf16.h>
#include <math.h>

// ---------------------------------------------------------------------------
// GRU language model forward (3-layer GRU, B=4, T=512, n=256, V=50257)
//   1) pack Wih/Whh/Wout fp32 -> bf16 WMMA-fragment-major layout
//   2) embedding gather -> bf16 [B*T, 256]
//   3) sequential GRU, one 16-wave workgroup, v_wmma_f32_16x16x32_bf16
//   4) logits GEMM [2048,256]x[256,50257], WMMA bf16, TDM tensor_load A-tiles
//   5) online logsumexp + NLL loss
// ---------------------------------------------------------------------------

typedef unsigned short u16;
typedef unsigned int   u32;

typedef __attribute__((ext_vector_type(16))) __bf16 v16bf;
typedef __attribute__((ext_vector_type(8)))  float  v8f;
typedef __attribute__((ext_vector_type(4)))  unsigned int v4u;
typedef __attribute__((ext_vector_type(8)))  int v8i;
typedef __attribute__((ext_vector_type(4)))  int v4i;

struct alignas(16) U128 { u32 x[4]; };
union Frag { U128 q[2]; v16bf v; u16 h[16]; };

#define T_SEQ 512
#define NSTATE 256
#define NGATE 768            // 3*NSTATE
#define VOCAB 50257
#define NT_V 3142            // ceil(VOCAB/16)

__device__ __forceinline__ u16 f2bf(float f) {
  u32 u = __float_as_uint(f);
  u32 r = u + 0x7FFFu + ((u >> 16) & 1u);   // round-to-nearest-even
  return (u16)(r >> 16);
}

// B-fragment (packed-weight) load: lane's 16 halves are contiguous (32B).
__device__ __forceinline__ v16bf load_frag16(const u16* p) {
  Frag f;
  f.q[0] = *(const U128*)(p);
  f.q[1] = *(const U128*)(p + 8);
  return f.v;
}

// A-fragment from row-major [16][256] bf16 (all 16 rows valid).
// ISA 16-bit A layout: lanes 0-15: M=lane, K {0..7,16..23}; lanes 16-31: M=lane-16, K {8..15,24..31}
__device__ __forceinline__ v16bf load_a16(const u16* base, int kt, int lane) {
  int row = lane & 15;
  const u16* p = base + row * NSTATE + kt * 32 + ((lane & 16) ? 8 : 0);
  Frag f;
  f.q[0] = *(const U128*)(p);
  f.q[1] = *(const U128*)(p + 16);
  return f.v;
}

// A-fragment with only rows 0..3 real: branchless — padding rows read a
// 64B zero scratchpad in LDS (address select -> v_cndmask, loads unconditional).
__device__ __forceinline__ v16bf load_a4z(const u16* base, const u16* zbuf,
                                          int kt, int lane) {
  int row = lane & 15;
  const u16* p = (row < 4)
      ? (base + row * NSTATE + kt * 32 + ((lane & 16) ? 8 : 0))
      : zbuf;
  Frag f;
  f.q[0] = *(const U128*)(p);
  f.q[1] = *(const U128*)(p + 16);
  return f.v;
}

// --------------------------- phase 1: weight packing ------------------------
// Fragment element (lane, h) <-> (kr, nr):
//   nr = lane & 15 ;  kr = (lane>=16 ? 8 : 0) + (h & 7) + (h>=8 ? 16 : 0)

__global__ void pack_rnn_w_kernel(const float* __restrict__ Wih,
                                  const float* __restrict__ Whh,
                                  u16* __restrict__ WihP,
                                  u16* __restrict__ WhhP) {
  int id = blockIdx.x * 256 + threadIdx.x;      // 0 .. 2*589824-1
  int which = (id >= 589824) ? 1 : 0;
  int rem = id - which * 589824;
  int f = rem >> 9;                              // l*384 + nt*8 + kt
  int e = rem & 511;
  int l  = f / 384;
  int fr = f - l * 384;
  int nt = fr >> 3, kt = fr & 7;
  int lane = e >> 4, h = e & 15;
  int nr = lane & 15;
  int kr = ((lane >> 4) << 3) + (h & 7) + ((h >> 3) << 4);
  int n = nt * 16 + nr;
  int k = kt * 32 + kr;
  const float* W = which ? Whh : Wih;
  u16* P = which ? WhhP : WihP;
  P[rem] = f2bf(W[((size_t)l * NSTATE + k) * NGATE + n]);
}

__global__ void pack_wout_kernel(const float* __restrict__ Wout,
                                 u16* __restrict__ WoutP) {
  int id = blockIdx.x * 256 + threadIdx.x;      // 0 .. 3142*8*512-1
  int f = id >> 9, e = id & 511;
  int nt = f >> 3, kt = f & 7;
  int lane = e >> 4, h = e & 15;
  int nr = lane & 15;
  int kr = ((lane >> 4) << 3) + (h & 7) + ((h >> 3) << 4);
  int n = nt * 16 + nr;
  int k = kt * 32 + kr;
  float v = (n < VOCAB) ? Wout[(size_t)k * VOCAB + n] : 0.0f;
  WoutP[id] = f2bf(v);
}

// --------------------------- phase 2: embedding -----------------------------
__global__ void embed_kernel(const int* __restrict__ idx,
                             const float* __restrict__ emb,
                             u16* __restrict__ Abf) {
  int row = blockIdx.x;                          // b*T + t
  int c = threadIdx.x;
  int tok = idx[row];
  Abf[(size_t)row * NSTATE + c] = f2bf(emb[(size_t)tok * NSTATE + c]);
}

// --------------------------- phase 3: GRU recurrence ------------------------
// One workgroup, 512 threads = 16 waves (4/SIMD -> big VGPR budget, no spills).
// Per layer: 96 WMMA N-tiles (48 x@Wih + 48 h@Whh) -> 6 tile-jobs per wave.
__global__ __launch_bounds__(512, 1)
void gru_kernel(const u16* __restrict__ Abf,
                const u16* __restrict__ WihP,
                const u16* __restrict__ WhhP,
                const float* __restrict__ bih,
                const float* __restrict__ bhh,
                u16* __restrict__ hidA) {
  __shared__ __align__(16) u16   x_lds[4 * NSTATE];      // layer-0 input bf16
  __shared__ __align__(16) u16   hb_lds[3 * 4 * NSTATE]; // bf16 hidden per layer
  __shared__ __align__(16) u16   zbuf[32];               // zero pad rows source
  __shared__ float hf_lds[3 * 4 * NSTATE];               // fp32 hidden per layer
  __shared__ float gi_lds[4 * NGATE];
  __shared__ float gh_lds[4 * NGATE];

  int tid = threadIdx.x;
  int lane = tid & 31;
  int w = tid >> 5;

  for (int i = tid; i < 3 * 4 * NSTATE; i += 512) { hf_lds[i] = 0.0f; hb_lds[i] = 0; }
  if (tid < 32) zbuf[tid] = 0;
  __syncthreads();

  for (int t = 0; t < T_SEQ; ++t) {
    // stage x_t (4 rows x 512B)
    if (tid < 128) {
      int b = tid >> 5, c = tid & 31;
      ((U128*)x_lds)[tid] =
          *((const U128*)(Abf + ((size_t)b * T_SEQ + t) * NSTATE) + c);
    }
    __syncthreads();

    for (int l = 0; l < 3; ++l) {
      #pragma unroll 1
      for (int j = 0; j < 6; ++j) {
        int job = w * 6 + j;                      // 0..95
        bool isIH = job < 48;
        int nt = isIH ? job : (job - 48);
        const u16* Abase = isIH ? (l == 0 ? x_lds : hb_lds + (l - 1) * 4 * NSTATE)
                                : (hb_lds + l * 4 * NSTATE);
        const u16* Bp = (isIH ? WihP : WhhP) +
                        ((size_t)(l * 48 + nt) * 8) * 512 + lane * 16;
        v8f acc = {0.f, 0.f, 0.f, 0.f, 0.f, 0.f, 0.f, 0.f};
        #pragma unroll
        for (int kt = 0; kt < 8; ++kt) {
          v16bf a = load_a4z(Abase, zbuf, kt, lane);
          v16bf b = load_frag16(Bp);
          acc = __builtin_amdgcn_wmma_f32_16x16x32_bf16(
              false, a, false, b, (short)0, acc, false, false);
          Bp += 512;
        }
        // C/D layout: lanes 0-15, VGPR r holds M=r; only rows 0..3 are real.
        if (lane < 16) {
          int n = nt * 16 + lane;
          float* dst = isIH ? gi_lds : gh_lds;
          const float* bias = (isIH ? bih : bhh) + l * NGATE;
          #pragma unroll
          for (int r = 0; r < 4; ++r) dst[r * NGATE + n] = acc[r] + bias[n];
        }
      }
      __syncthreads();

      // elementwise gates: (b, jj) over 4x256
      for (int i = tid; i < 4 * NSTATE; i += 512) {
        int b = i >> 8, jj = i & 255;
        float gr = gi_lds[b * NGATE + jj] + gh_lds[b * NGATE + jj];
        float gz = gi_lds[b * NGATE + 256 + jj] + gh_lds[b * NGATE + 256 + jj];
        float ig = gi_lds[b * NGATE + 512 + jj];
        float hg = gh_lds[b * NGATE + 512 + jj];
        float rr = 1.0f / (1.0f + __expf(-gr));
        float zz = 1.0f / (1.0f + __expf(-gz));
        float gg = tanhf(ig + rr * hg);
        float hp = hf_lds[(l * 4 + b) * NSTATE + jj];
        float hn = (1.0f - zz) * gg + zz * hp;
        hf_lds[(l * 4 + b) * NSTATE + jj] = hn;
        u16 hb = f2bf(hn);
        hb_lds[(l * 4 + b) * NSTATE + jj] = hb;
        if (l == 2) hidA[((size_t)b * T_SEQ + t) * NSTATE + jj] = hb;
      }
      __syncthreads();
    }
  }
}

// --------------------------- phase 4: logits GEMM ---------------------------
// Block = 8 waves; block covers 32 N-tiles of one M-tile; each wave caches the
// 8 A-fragments in registers and sweeps 4 N-tiles. A-tile staged via TDM.
__global__ __launch_bounds__(256)
void gemm_kernel(const u16* __restrict__ hidA,
                 const u16* __restrict__ WoutP,
                 const float* __restrict__ bout,
                 float* __restrict__ out) {
  __shared__ __align__(16) u16 As[16 * NSTATE];
  int tid = threadIdx.x;
  int lane = tid & 31;
  int w = tid >> 5;
  int mtile = blockIdx.y;

#if __has_builtin(__builtin_amdgcn_tensor_load_to_lds) && __has_builtin(__builtin_amdgcn_s_wait_tensorcnt)
  if (w == 0) {
    // Tensor Data Mover: 2-D tile, 256 x 16 elements, 2B each, stride 256.
    unsigned long long ga =
        (unsigned long long)(const void*)(hidA + (size_t)mtile * 16 * NSTATE);
    unsigned lds = (unsigned)(unsigned long long)(const void*)As;  // LDS byte offset
    v4u g0;
    g0.x = 1u;                                       // count=1 (valid user D#)
    g0.y = lds;                                      // lds_addr
    g0.z = (u32)(ga & 0xFFFFFFFFu);                  // global_addr[31:0]
    g0.w = (u32)((ga >> 32) & 0x01FFFFFFu) | (2u << 30);  // addr[56:32] | type=2
    v8i g1;
    g1[0] = 0x00010000;                              // data_size=1 (2 bytes)
    g1[1] = (int)(256u << 16);                       // tensor_dim0 = 256
    g1[2] = (int)(16u << 16);                        // tensor_dim1 = 16
    g1[3] = (int)(256u << 16);                       // tile_dim0 = 256
    g1[4] = 16;                                      // tile_dim1 = 16, tile_dim2 = 0
    g1[5] = 256;                                     // tensor_dim0_stride = 256
    g1[6] = 0;
    g1[7] = 0;
    v4i g2 = {0, 0, 0, 0};
    v4i g3 = {0, 0, 0, 0};
    v8i g4 = {0, 0, 0, 0, 0, 0, 0, 0};               // 6-arg form: extra group
    __builtin_amdgcn_tensor_load_to_lds(g0, g1, g2, g3, g4, 0);
    __builtin_amdgcn_s_wait_tensorcnt(0);
  }
  __syncthreads();
#else
  {   // fallback: cooperative 8KB copy
    const U128* src = (const U128*)(hidA + (size_t)mtile * 16 * NSTATE);
    U128* dst = (U128*)As;
    dst[tid]       = src[tid];
    dst[tid + 256] = src[tid + 256];
  }
  __syncthreads();
#endif

  // cache all 8 A-fragments in registers
  v16bf afrag[8];
  #pragma unroll
  for (int kt = 0; kt < 8; ++kt) afrag[kt] = load_a16(As, kt, lane);

  int ntile0 = blockIdx.x * 32 + w * 4;
  #pragma unroll 1
  for (int q = 0; q < 4; ++q) {
    int ntile = ntile0 + q;
    if (ntile >= NT_V) break;                     // wave-uniform
    const u16* Bp = WoutP + (size_t)ntile * 8 * 512 + lane * 16;
    v8f acc = {0.f, 0.f, 0.f, 0.f, 0.f, 0.f, 0.f, 0.f};
    #pragma unroll
    for (int kt = 0; kt < 8; ++kt) {
      __builtin_prefetch(Bp + 512, 0, 1);         // global_prefetch_b8, next frag
      v16bf b = load_frag16(Bp);
      acc = __builtin_amdgcn_wmma_f32_16x16x32_bf16(
          false, afrag[kt], false, b, (short)0, acc, false, false);
      Bp += 512;
    }
    int n = ntile * 16 + (lane & 15);
    if (n < VOCAB) {
      float bias = bout[n];
      int mbase = mtile * 16 + ((lane & 16) ? 8 : 0);
      #pragma unroll
      for (int r = 0; r < 8; ++r)
        out[(size_t)(mbase + r) * VOCAB + n] = acc[r] + bias;
    }
  }
}

// --------------------------- phase 5: loss ----------------------------------
__global__ void zero_loss_kernel(float* loss) { *loss = 0.0f; }

__global__ __launch_bounds__(256)
void loss_kernel(const float* __restrict__ logits,
                 const int* __restrict__ targets,
                 float* __restrict__ loss) {
  __shared__ float sm[256], ss[256];
  int m = blockIdx.x;
  int tid = threadIdx.x;
  const float* row = logits + (size_t)m * VOCAB;

  float lm = -INFINITY, ls = 0.0f;               // online logsumexp
  for (int v = tid; v < VOCAB; v += 256) {
    float x = row[v];
    if (x > lm) { ls = ls * __expf(lm - x) + 1.0f; lm = x; }
    else        { ls += __expf(x - lm); }
  }
  sm[tid] = lm; ss[tid] = ls;
  __syncthreads();
  for (int o = 128; o > 0; o >>= 1) {
    if (tid < o) {
      float m2 = sm[tid + o], s2 = ss[tid + o];
      float M = fmaxf(sm[tid], m2);
      ss[tid] = ss[tid] * __expf(sm[tid] - M) + s2 * __expf(m2 - M);
      sm[tid] = M;
    }
    __syncthreads();
  }
  if (tid == 0) {
    float lt = row[targets[m]];
    float lp = lt - sm[0] - __logf(ss[0]);
    atomicAdd(loss, -lp * (1.0f / 2048.0f));
  }
}

// --------------------------- launcher ---------------------------------------
extern "C" void kernel_launch(void* const* d_in, const int* in_sizes, int n_in,
                              void* d_out, int out_size, void* d_ws, size_t ws_size,
                              hipStream_t stream) {
  (void)in_sizes; (void)n_in; (void)ws_size;
  const int*   idx  = (const int*)  d_in[0];
  const int*   tgt  = (const int*)  d_in[1];
  const float* emb  = (const float*)d_in[2];
  const float* Wih  = (const float*)d_in[3];
  const float* Whh  = (const float*)d_in[4];
  const float* bih  = (const float*)d_in[5];
  const float* bhh  = (const float*)d_in[6];
  const float* Wout = (const float*)d_in[7];
  const float* bout = (const float*)d_in[8];
  float* out = (float*)d_out;

  // workspace layout (bytes)
  char* ws = (char*)d_ws;
  u16* Abf   = (u16*)(ws + 0);              // 2048*256*2        = 1,048,576
  u16* WihP  = (u16*)(ws + 1048576);        // 3*48*8*512*2      = 1,179,648
  u16* WhhP  = (u16*)(ws + 2228224);        // 1,179,648
  u16* hidA  = (u16*)(ws + 3407872);        // 2048*256*2        = 1,048,576
  u16* WoutP = (u16*)(ws + 4456448);        // 3142*8*512*2      = 25,739,264

  pack_rnn_w_kernel<<<4608, 256, 0, stream>>>(Wih, Whh, WihP, WhhP);
  pack_wout_kernel<<<50272, 256, 0, stream>>>(Wout, WoutP);
  embed_kernel<<<2048, 256, 0, stream>>>(idx, emb, Abf);
  gru_kernel<<<1, 512, 0, stream>>>(Abf, WihP, WhhP, bih, bhh, hidA);
  gemm_kernel<<<dim3((NT_V + 31) / 32, 128), 256, 0, stream>>>(hidA, WoutP, bout, out);
  zero_loss_kernel<<<1, 1, 0, stream>>>(out + (out_size - 1));
  loss_kernel<<<2048, 256, 0, stream>>>(out, tgt, out + (out_size - 1));
}
